// SimpleLSTM_71811853189158
// MI455X (gfx1250) — matrix operand order repaired and checked by reading, not compiled, not run on hardware
//
#include <hip/hip_runtime.h>
#include <hip/hip_bf16.h>
#include <stdint.h>

#define INPUT_D 100
#define KPAD    128
#define HIDDEN  64
#define G4      256
#define T_LEN   512
#define B_SZ    1024
#define ROWS    16       // batch rows per workgroup (WMMA M)
#define NTHREADS 256     // 8 waves of 32
#define W0_LDK  136      // padded bf16 row stride for Wih0 in LDS (bank-conflict-free)
#define H_LDK   72       // padded bf16 row stride for h buffers in LDS

typedef __attribute__((ext_vector_type(16))) __bf16 v16bf;
typedef __attribute__((ext_vector_type(8)))  float  v8f;

union Frag {
    v16bf v;
    uint32_t u32[8];
};

static __device__ __forceinline__ uint32_t pack2bf(float a, float b) {
    union { __bf16 h[2]; uint32_t u; } c;
    c.h[0] = (__bf16)a;          // native v_cvt (pk) bf16
    c.h[1] = (__bf16)b;
    return c.u;
}
static __device__ __forceinline__ float sigm(float x) { return 1.0f / (1.0f + __expf(-x)); }
static __device__ __forceinline__ float tanh_fast(float x) {
    float e = __expf(2.0f * x);
    return 1.0f - 2.0f / (e + 1.0f);
}

__global__ __launch_bounds__(NTHREADS, 1)
void lstm2_fused_wmma(const float* __restrict__ x,
                      const float* __restrict__ Wih0, const float* __restrict__ Whh0,
                      const float* __restrict__ bih0, const float* __restrict__ bhh0,
                      const float* __restrict__ Wih1, const float* __restrict__ Whh1,
                      const float* __restrict__ bih1, const float* __restrict__ bhh1,
                      const float* __restrict__ fcw,  const float* __restrict__ fcb,
                      float* __restrict__ out)
{
    // ~99 KB total LDS (<< 320 KB/WGP)
    __shared__ __bf16 s_wih0[G4 * W0_LDK];           // 68 KB: bf16 Wih0, K padded 100->128(+8)
    __shared__ float  s_gates[ROWS][G4];             // 16 KB
    __shared__ float  s_c0[ROWS][HIDDEN];            // 4 KB
    __shared__ float  s_c1[ROWS][HIDDEN];            // 4 KB
    __shared__ __bf16 s_h0[ROWS][H_LDK];             // 2.25 KB
    __shared__ __bf16 s_h1[ROWS][H_LDK];             // 2.25 KB
    __shared__ float  s_fcw[HIDDEN];

    const int tid = threadIdx.x;
    const int w   = tid >> 5;    // wave id 0..7 -> gate columns [32w, 32w+32)
    const int l   = tid & 31;    // lane in wave32
    const int lm  = l & 15;      // A: row m   |  B/C: col n
    const int lh  = l >> 4;      // lane half
    const int r0  = blockIdx.x * ROWS;

    if (tid < HIDDEN) s_fcw[tid] = fcw[tid];
    #pragma unroll
    for (int s = 0; s < (ROWS * HIDDEN) / NTHREADS; ++s) {
        int idx = tid + NTHREADS * s;
        int m = idx >> 6, j = idx & 63;
        s_c0[m][j] = 0.0f;
        s_c1[m][j] = 0.0f;
    }
    // one-time: Wih0 -> LDS as bf16, layout [n][k] with padded stride.
    // thread tid owns gate-row n = tid (G4 == NTHREADS).
    {
        const float* w0 = Wih0 + tid * INPUT_D;
        __bf16* dst = s_wih0 + tid * W0_LDK;
        #pragma unroll 4
        for (int k = 0; k < KPAD; ++k)
            dst[k] = (k < INPUT_D) ? (__bf16)w0[k] : (__bf16)0.0f;
    }
    const float fcb0 = fcb[0];

    // ---- one-time: recurrent-path weight B-fragments (bf16) in registers ----
    // B 32x16 layout: lane holds col n = l&15; element j -> k = 32*kt + 16*lh + j
    Frag whh0f[2][2], wih1f[2][2], whh1f[2][2];
    float bia0[2], bia1[2];

    #pragma unroll
    for (int nt = 0; nt < 2; ++nt) {
        const int n = 32 * w + 16 * nt + lm;
        bia0[nt] = bih0[n] + bhh0[n];
        bia1[nt] = bih1[n] + bhh1[n];
        const float* wh0 = Whh0 + n * HIDDEN;
        const float* wi1 = Wih1 + n * HIDDEN;
        const float* wh1 = Whh1 + n * HIDDEN;
        #pragma unroll
        for (int kt = 0; kt < 2; ++kt) {
            const int kb = 32 * kt + 16 * lh;
            #pragma unroll
            for (int e = 0; e < 8; ++e) {
                whh0f[nt][kt].u32[e] = pack2bf(wh0[kb + 2*e], wh0[kb + 2*e + 1]);
                wih1f[nt][kt].u32[e] = pack2bf(wi1[kb + 2*e], wi1[kb + 2*e + 1]);
                whh1f[nt][kt].u32[e] = pack2bf(wh1[kb + 2*e], wh1[kb + 2*e + 1]);
            }
        }
    }

    // hidden-state A-fragments (h = 0 at t = 0)
    Frag h0f[2], h1f[2];
    #pragma unroll
    for (int kt = 0; kt < 2; ++kt)
        #pragma unroll
        for (int e = 0; e < 8; ++e) { h0f[kt].u32[e] = 0u; h1f[kt].u32[e] = 0u; }

    __syncthreads();

    const float* xrow = x + (size_t)(r0 + lm) * T_LEN * INPUT_D;

    for (int t = 0; t < T_LEN; ++t) {
        const float* xp = xrow + (size_t)t * INPUT_D;
        __builtin_prefetch(xp + INPUT_D, 0, 1);   // next timestep -> global_prefetch

        // ---- build x A-fragments (bf16): A 16x32, k = 32*kt + 16*lh + j
        Frag xf[4];
        #pragma unroll
        for (int kt = 0; kt < 4; ++kt) {
            #pragma unroll
            for (int h2 = 0; h2 < 2; ++h2) {
                const int kb = 32 * kt + 16 * lh + 8 * h2;
                #pragma unroll
                for (int e = 0; e < 4; ++e) {
                    float a, b;
                    if (kt < 3) { a = xp[kb + 2*e]; b = xp[kb + 2*e + 1]; }
                    else {
                        int k0 = kb + 2 * e;
                        a = (k0     < INPUT_D) ? xp[k0]     : 0.0f;
                        b = (k0 + 1 < INPUT_D) ? xp[k0 + 1] : 0.0f;
                    }
                    xf[kt].u32[h2 * 4 + e] = pack2bf(a, b);
                }
            }
        }

        // ---- layer 0: gates = x @ Wih0^T + h0 @ Whh0^T + (b_ih0 + b_hh0) ----
        #pragma unroll
        for (int nt = 0; nt < 2; ++nt) {
            const int n = 32 * w + 16 * nt + lm;
            v8f acc;
            #pragma unroll
            for (int r = 0; r < 8; ++r) acc[r] = bia0[nt];
            #pragma unroll
            for (int kt = 0; kt < 4; ++kt) {
                // stream Wih0 B-fragment from LDS (2 x ds_load_b128 per lane)
                Frag wf;
                const uint32_t* p =
                    (const uint32_t*)(s_wih0 + n * W0_LDK + 32 * kt + 16 * lh);
                #pragma unroll
                for (int e = 0; e < 8; ++e) wf.u32[e] = p[e];
                acc = __builtin_amdgcn_wmma_f32_16x16x32_bf16(
                        false, xf[kt].v, false, wf.v, (short)0, acc, false, false);
            }
            #pragma unroll
            for (int kt = 0; kt < 2; ++kt)
                acc = __builtin_amdgcn_wmma_f32_16x16x32_bf16(
                        false, h0f[kt].v, false, whh0f[nt][kt].v, (short)0, acc, false, false);
            #pragma unroll
            for (int r = 0; r < 8; ++r) s_gates[8 * lh + r][n] = acc[r];   // C: n=l&15, m=r+8*lh
        }
        __syncthreads();

        // ---- elementwise LSTM cell, layer 0 ----
        #pragma unroll
        for (int s = 0; s < 4; ++s) {
            int idx = tid + NTHREADS * s;
            int m = idx >> 6, j = idx & 63;
            float ig = sigm(s_gates[m][j]);
            float fg = sigm(s_gates[m][64 + j]);
            float gg = tanh_fast(s_gates[m][128 + j]);
            float og = sigm(s_gates[m][192 + j]);
            float c  = fmaf(fg, s_c0[m][j], ig * gg);
            s_c0[m][j] = c;
            s_h0[m][j] = (__bf16)(og * tanh_fast(c));
        }
        __syncthreads();

        // ---- rebuild h0 A-fragments from LDS (aligned 16B reads, padded rows) ----
        #pragma unroll
        for (int kt = 0; kt < 2; ++kt) {
            #pragma unroll
            for (int h2 = 0; h2 < 2; ++h2) {
                const int kb = 32 * kt + 16 * lh + 8 * h2;
                const uint32_t* p = (const uint32_t*)&s_h0[lm][kb];
                #pragma unroll
                for (int e = 0; e < 4; ++e) h0f[kt].u32[h2 * 4 + e] = p[e];
            }
        }

        // ---- layer 1: gates = h0 @ Wih1^T + h1 @ Whh1^T + (b_ih1 + b_hh1) ----
        #pragma unroll
        for (int nt = 0; nt < 2; ++nt) {
            v8f acc;
            #pragma unroll
            for (int r = 0; r < 8; ++r) acc[r] = bia1[nt];
            #pragma unroll
            for (int kt = 0; kt < 2; ++kt)
                acc = __builtin_amdgcn_wmma_f32_16x16x32_bf16(
                        false, h0f[kt].v, false, wih1f[nt][kt].v, (short)0, acc, false, false);
            #pragma unroll
            for (int kt = 0; kt < 2; ++kt)
                acc = __builtin_amdgcn_wmma_f32_16x16x32_bf16(
                        false, h1f[kt].v, false, whh1f[nt][kt].v, (short)0, acc, false, false);
            const int n = 32 * w + 16 * nt + lm;
            #pragma unroll
            for (int r = 0; r < 8; ++r) s_gates[8 * lh + r][n] = acc[r];
        }
        __syncthreads();

        // ---- elementwise LSTM cell, layer 1 ----
        #pragma unroll
        for (int s = 0; s < 4; ++s) {
            int idx = tid + NTHREADS * s;
            int m = idx >> 6, j = idx & 63;
            float ig = sigm(s_gates[m][j]);
            float fg = sigm(s_gates[m][64 + j]);
            float gg = tanh_fast(s_gates[m][128 + j]);
            float og = sigm(s_gates[m][192 + j]);
            float c  = fmaf(fg, s_c1[m][j], ig * gg);
            s_c1[m][j] = c;
            s_h1[m][j] = (__bf16)(og * tanh_fast(c));
        }
        __syncthreads();

        // ---- rebuild h1 A-fragments ----
        #pragma unroll
        for (int kt = 0; kt < 2; ++kt) {
            #pragma unroll
            for (int h2 = 0; h2 < 2; ++h2) {
                const int kb = 32 * kt + 16 * lh + 8 * h2;
                const uint32_t* p = (const uint32_t*)&s_h1[lm][kb];
                #pragma unroll
                for (int e = 0; e < 4; ++e) h1f[kt].u32[h2 * 4 + e] = p[e];
            }
        }

        // ---- final 64 -> 1 projection for this timestep ----
        if (tid < ROWS) {
            float sum = fcb0;
            #pragma unroll
            for (int j = 0; j < HIDDEN; ++j)
                sum = fmaf((float)s_h1[tid][j], s_fcw[j], sum);
            out[(size_t)(r0 + tid) * T_LEN + t] = sum;
        }
    }
}

extern "C" void kernel_launch(void* const* d_in, const int* in_sizes, int n_in,
                              void* d_out, int out_size, void* d_ws, size_t ws_size,
                              hipStream_t stream) {
    (void)in_sizes; (void)n_in; (void)out_size; (void)d_ws; (void)ws_size;
    const float* x    = (const float*)d_in[0];
    const float* Wih0 = (const float*)d_in[1];
    const float* Whh0 = (const float*)d_in[2];
    const float* bih0 = (const float*)d_in[3];
    const float* bhh0 = (const float*)d_in[4];
    const float* Wih1 = (const float*)d_in[5];
    const float* Whh1 = (const float*)d_in[6];
    const float* bih1 = (const float*)d_in[7];
    const float* bhh1 = (const float*)d_in[8];
    const float* fcw  = (const float*)d_in[9];
    const float* fcb  = (const float*)d_in[10];
    float* out = (float*)d_out;

    dim3 grid(B_SZ / ROWS);   // 64 workgroups, each owns 16 batch rows for all T
    dim3 block(NTHREADS);     // 8 wave32s: each owns 32 of the 256 gate columns
    lstm2_fused_wmma<<<grid, block, 0, stream>>>(
        x, Wih0, Whh0, bih0, bhh0, Wih1, Whh1, bih1, bhh1, fcw, fcb, out);
}
